// IDDecoder_42374147342993
// MI455X (gfx1250) — matrix operand order repaired and checked by reading, not compiled, not run on hardware
//
#include <hip/hip_runtime.h>

#define NTOK   1280   // 64*20 tokens
#define AD     512
#define NH     16
#define HD     32
#define FFN    2048
#define VOC    101
#define HID    256
#define MAXT   30
#define TT     20
#define SROW   40     // padded LDS row stride (bf16 elems) for 32-K tiles

typedef __attribute__((ext_vector_type(16))) __bf16 v16bf;
typedef __attribute__((ext_vector_type(8)))  __bf16 v8bf;
typedef __attribute__((ext_vector_type(8)))  float  v8f;
typedef __attribute__((ext_vector_type(4)))  float  v4f;
typedef __attribute__((ext_vector_type(4)))  int    v4i_t;

// ---------- CDNA5 async global->LDS copy (ASYNCcnt path), with sync fallback ----------
#if __has_builtin(__builtin_amdgcn_global_load_async_to_lds_b128)
#define HAS_ASYNC_LDS 1
#else
#define HAS_ASYNC_LDS 0
#endif

__device__ __forceinline__ void cp16_to_lds(void* lds_dst, const void* gsrc) {
#if HAS_ASYNC_LDS
    __builtin_amdgcn_global_load_async_to_lds_b128(
        (__attribute__((address_space(1))) v4i_t*)gsrc,
        (__attribute__((address_space(3))) v4i_t*)lds_dst, 0, 0);
#else
    *(uint4*)lds_dst = *(const uint4*)gsrc;
#endif
}

__device__ __forceinline__ void cp_lds_wait() {
#if HAS_ASYNC_LDS
#if __has_builtin(__builtin_amdgcn_s_wait_asynccnt)
    __builtin_amdgcn_s_wait_asynccnt(0);
#else
    asm volatile("s_wait_asynccnt 0" ::: "memory");
#endif
#endif
}

// ---------- fragment loaders (wave32 WMMA layouts, CDNA5 ISA) ----------
__device__ __forceinline__ v16bf frag_from_2x8(const __bf16* p0, const __bf16* p1) {
    v8bf lo = *(const v8bf*)p0;
    v8bf hi = *(const v8bf*)p1;
    v16bf r;
#pragma unroll
    for (int i = 0; i < 8; i++) { r[i] = lo[i]; r[8 + i] = hi[i]; }
    return r;
}

// A-matrix 16x32: lane (half*16+m): 16 contiguous K values half*16..
__device__ __forceinline__ v16bf load_a_frag(const __bf16* row_k0, int half) {
    return frag_from_2x8(row_k0 + half * 16, row_k0 + half * 16 + 8);
}

// B-matrix 32x16 from bf16 memory: lane (half*16+n): 16 contiguous K values.
__device__ __forceinline__ v16bf load_b_frag_bf16(const __bf16* p) {
    return frag_from_2x8(p, p + 8);
}

// B-matrix 32x16 from f32 weights (convert on the fly).
__device__ __forceinline__ v16bf load_b_frag_f32(const float* wp) {
    v16bf r;
#pragma unroll
    for (int q = 0; q < 4; q++) {
        v4f t = *(const v4f*)(wp + q * 4);
#pragma unroll
        for (int i = 0; i < 4; i++) r[q * 4 + i] = (__bf16)t[i];
    }
    return r;
}

#define WMMA_BF16(a, b, c) \
    __builtin_amdgcn_wmma_f32_16x16x32_bf16(false, (a), false, (b), (short)0, (c), false, false)

// ---------- f32 -> bf16 weight conversion ----------
__global__ void f32_to_bf16_kernel(const float* __restrict__ src, __bf16* __restrict__ dst,
                                   int n) {
    int i = blockIdx.x * blockDim.x + threadIdx.x;
    if (i < n) dst[i] = (__bf16)src[i];
}

// ---------- input build ----------
__global__ void build_inputs_kernel(const float* __restrict__ tfe, const int* __restrict__ tids,
                                    const float* __restrict__ uf, const int* __restrict__ uids,
                                    const float* __restrict__ w2e,
                                    __bf16* __restrict__ trajb, float* __restrict__ xf,
                                    __bf16* __restrict__ xb) {
    int idx = blockIdx.x * blockDim.x + threadIdx.x;
    if (idx >= NTOK * AD) return;
    int row = idx >> 9, col = idx & (AD - 1);
    float tv, uv;
    if (col < HID) {
        tv = tfe[row * HID + col];
        uv = uf[row * HID + col];
    } else {
        int c = col - HID;
        tv = w2e[c * VOC + tids[row]];
        uv = w2e[c * VOC + uids[row]];
    }
    trajb[idx] = (__bf16)tv;
    xf[idx] = uv;
    xb[idx] = (__bf16)uv;
}

// ---------- main WMMA GEMM: 64x64 tile per workgroup (4 waves), LDS double-buffered ----------
// C[M x N] = act(A[M x K](bf16) @ W[N x K](bf16)^T + bias); M,N multiples of 64.
__global__ __launch_bounds__(128) void gemm_wmma64_kernel(
    const __bf16* __restrict__ A, int lda, const __bf16* __restrict__ W, int ldw,
    const float* __restrict__ bias, float* __restrict__ Cf, __bf16* __restrict__ Cb,
    int ldc, int K, int act) {
    __shared__ __bf16 sA[2][64 * SROW];
    __shared__ __bf16 sW[2][64 * SROW];

    const int tid  = threadIdx.x;
    const int wave = tid >> 5;
    const int lane = tid & 31;
    const int half = lane >> 4;
    const int l15  = lane & 15;
    const int m0   = blockIdx.x * 64;
    const int n0   = blockIdx.y * 64;

    // Stage one 64x32 tile of A and W into LDS buffer `buf` (each thread: 2x16B per matrix).
    auto stage = [&](int buf, int k0) {
        int c0 = tid * 2;
#pragma unroll
        for (int u = 0; u < 2; u++) {
            int c   = c0 + u;
            int row = c >> 2;           // 0..63
            int off = (c & 3) * 8;      // 0,8,16,24 (bf16 elems)
            cp16_to_lds(&sA[buf][row * SROW + off], A + (size_t)(m0 + row) * lda + k0 + off);
            cp16_to_lds(&sW[buf][row * SROW + off], W + (size_t)(n0 + row) * ldw + k0 + off);
        }
    };

    v8f acc0 = {}, acc1 = {}, acc2 = {}, acc3 = {};
    const int nk = K >> 5;

    stage(0, 0);
    for (int i = 0; i < nk; i++) {
        cp_lds_wait();        // own async tile writes done
        __syncthreads();      // everyone's writes done + prev compute finished
        if (i + 1 < nk) stage((i + 1) & 1, (i + 1) * 32);

        const __bf16* pa = &sA[i & 1][(wave * 16 + l15) * SROW];
        v16bf a = frag_from_2x8(pa + half * 16, pa + half * 16 + 8);

        const __bf16* pw = &sW[i & 1][0];
        v16bf b0 = load_b_frag_bf16(pw + (0 * 16 + l15) * SROW + half * 16);
        v16bf b1 = load_b_frag_bf16(pw + (1 * 16 + l15) * SROW + half * 16);
        v16bf b2 = load_b_frag_bf16(pw + (2 * 16 + l15) * SROW + half * 16);
        v16bf b3 = load_b_frag_bf16(pw + (3 * 16 + l15) * SROW + half * 16);

        acc0 = WMMA_BF16(a, b0, acc0);
        acc1 = WMMA_BF16(a, b1, acc1);
        acc2 = WMMA_BF16(a, b2, acc2);
        acc3 = WMMA_BF16(a, b3, acc3);
    }

    // epilogue
    v8f* accs[4] = {&acc0, &acc1, &acc2, &acc3};
#pragma unroll
    for (int nt = 0; nt < 4; nt++) {
        int n = n0 + nt * 16 + l15;
        float bv = bias ? bias[n] : 0.0f;
        v8f acc = *accs[nt];
#pragma unroll
        for (int r = 0; r < 8; r++) {
            int row = m0 + wave * 16 + r + 8 * half;
            float v = acc[r] + bv;
            if (act == 1) v = 0.5f * v * (1.0f + erff(v * 0.70710678118654752f));
            if (Cf) Cf[(size_t)row * ldc + n] = v;
            if (Cb) Cb[(size_t)row * ldc + n] = (__bf16)v;
        }
    }
}

// ---------- small WMMA GEMM (tail N, f32 weights): used for final 1280x101 ----------
__global__ __launch_bounds__(32) void gemm_wmma_small_kernel(
    const __bf16* __restrict__ A, int lda, const float* __restrict__ W, int ldw,
    const float* __restrict__ bias, float* __restrict__ Cf, __bf16* __restrict__ Cb,
    int ldc, int N, int K, int act) {
    const int m0   = blockIdx.x * 16;
    const int n0   = blockIdx.y * 16;
    const int lane = threadIdx.x;
    const int half = lane >> 4;
    const int l15  = lane & 15;

    const int m = m0 + l15;
    const int n = n0 + l15;
    const bool nv = (n < N);
    const __bf16* arow = A + (size_t)m * lda;
    const float*  wrow = W + (size_t)(nv ? n : 0) * ldw;

    v8f acc = {};
    for (int k0 = 0; k0 < K; k0 += 32) {
        v16bf a = load_a_frag(arow + k0, half);
        v16bf b;
        if (nv) {
            b = load_b_frag_f32(wrow + k0 + half * 16);
        } else {
#pragma unroll
            for (int i = 0; i < 16; i++) b[i] = (__bf16)0.0f;
        }
        if (k0 + 32 < K) {
            __builtin_prefetch(arow + k0 + 32, 0, 1);
            __builtin_prefetch(wrow + k0 + 32, 0, 1);
        }
        acc = WMMA_BF16(a, b, acc);
    }

    if (nv) {
        float bv = bias ? bias[n] : 0.0f;
#pragma unroll
        for (int r = 0; r < 8; r++) {
            int row = m0 + r + 8 * half;
            float v = acc[r] + bv;
            if (act == 1) v = 0.5f * v * (1.0f + erff(v * 0.70710678118654752f));
            if (Cf) Cf[(size_t)row * ldc + n] = v;
            if (Cb) Cb[(size_t)row * ldc + n] = (__bf16)v;
        }
    }
}

// ---------- fused flash attention, one wave per (16-query tile, head) ----------
// mode 0: cross attn (1280 traj keys, rel-time bias + block-causal mask)
// mode 1: self attn per time step t (64 agent keys, stride TT)
// Both fold in the appended zero "sink" key.
__global__ __launch_bounds__(32) void attn_fused_kernel(
    const __bf16* __restrict__ Q, const __bf16* __restrict__ Kb, const __bf16* __restrict__ Vb,
    __bf16* __restrict__ O, const int* __restrict__ qtimes, const int* __restrict__ ktimes,
    const float* __restrict__ rel, int mode, int nkb) {
    const int h    = blockIdx.y;
    const int lane = threadIdx.x;
    const int half = lane >> 4;
    const int l15  = lane & 15;

    int t = 0, qbase = 0;
    if (mode == 0) {
        qbase = blockIdx.x * 16;
    } else {
        t = blockIdx.x >> 2;
        qbase = (blockIdx.x & 3) * 16;
    }

    __shared__ float  s_s[16][36];
    __shared__ __bf16 s_p[16][32];
    __shared__ float  s_m[16], s_l[16], s_c[16];
    __shared__ int    s_qt[16];

    if (lane < 16) {
        s_m[lane] = -1e30f;
        s_l[lane] = 0.0f;
        int qg = (mode == 0) ? (qbase + lane) : ((qbase + lane) * TT + t);
        s_qt[lane] = (mode == 0) ? qtimes[qg] : 0;
    }

    const int qg0 = (mode == 0) ? (qbase + l15) : ((qbase + l15) * TT + t);
    v16bf qa = load_a_frag(Q + (size_t)qg0 * AD + h * HD, half);

    v8f o0 = {}, o1 = {};
    __syncthreads();

    const float scale = 0.17677669529663687f; // 1/sqrt(32)

    for (int kb = 0; kb < nkb; kb++) {
        const int j0 = kb * 32;
        int kf0 = (mode == 0) ? (j0 + l15) : ((j0 + l15) * TT + t);
        int kf1 = (mode == 0) ? (j0 + 16 + l15) : ((j0 + 16 + l15) * TT + t);

        v16bf b0 = load_b_frag_bf16(Kb + (size_t)kf0 * AD + h * HD + half * 16);
        v16bf b1 = load_b_frag_bf16(Kb + (size_t)kf1 * AD + h * HD + half * 16);

        v8f sc0 = {}, sc1 = {};
        sc0 = WMMA_BF16(qa, b0, sc0);
        sc1 = WMMA_BF16(qa, b1, sc1);

#pragma unroll
        for (int r = 0; r < 8; r++) {
            int row = r + 8 * half;
            float v0 = sc0[r] * scale;
            float v1 = sc1[r] * scale;
            if (mode == 0) {
                int qt = s_qt[row];
                int tq = (qbase + row) % TT;
                int ja = j0 + l15, jb = j0 + 16 + l15;
                int r0 = (qt - ktimes[ja]) % MAXT; if (r0 < 0) r0 += MAXT;
                int r1 = (qt - ktimes[jb]) % MAXT; if (r1 < 0) r1 += MAXT;
                v0 += rel[r0 * NH + h];
                v1 += rel[r1 * NH + h];
                if ((ja % TT) > tq) v0 = -1e30f;
                if ((jb % TT) > tq) v1 = -1e30f;
            }
            s_s[row][l15] = v0;
            s_s[row][16 + l15] = v1;
        }
        __syncthreads();

        if (lane < 16) {
            float mold = s_m[lane], mx = mold;
#pragma unroll
            for (int c = 0; c < 32; c++) mx = fmaxf(mx, s_s[lane][c]);
            float corr = __expf(mold - mx);
            float sum = 0.0f;
#pragma unroll
            for (int c = 0; c < 32; c++) {
                float p = __expf(s_s[lane][c] - mx);
                s_p[lane][c] = (__bf16)p;
                sum += p;
            }
            s_l[lane] = s_l[lane] * corr + sum;
            s_m[lane] = mx;
            s_c[lane] = corr;
        }
        __syncthreads();

#pragma unroll
        for (int r = 0; r < 8; r++) {
            float c = s_c[r + 8 * half];
            o0[r] *= c;
            o1[r] *= c;
        }

        v16bf pf;
#pragma unroll
        for (int i = 0; i < 8; i++) {
            pf[i]     = s_p[l15][half * 16 + i];
            pf[8 + i] = s_p[l15][half * 16 + 8 + i];
        }

        v16bf vf0, vf1;
#pragma unroll
        for (int i = 0; i < 16; i++) {
            int jk = j0 + half * 16 + i;
            int kf = (mode == 0) ? jk : (jk * TT + t);
            const __bf16* vp = Vb + (size_t)kf * AD + h * HD;
            vf0[i] = vp[l15];
            vf1[i] = vp[16 + l15];
        }
        o0 = WMMA_BF16(pf, vf0, o0);
        o1 = WMMA_BF16(pf, vf1, o1);
        __syncthreads();
    }

    if (lane < 16) {
        float l = s_l[lane] + __expf(0.0f - s_m[lane]);
        s_c[lane] = 1.0f / l;
    }
    __syncthreads();

#pragma unroll
    for (int r = 0; r < 8; r++) {
        int row = r + 8 * half;
        int qg = (mode == 0) ? (qbase + row) : ((qbase + row) * TT + t);
        float inv = s_c[row];
        __bf16* op = O + (size_t)qg * AD + h * HD;
        op[l15]      = (__bf16)(o0[r] * inv);
        op[16 + l15] = (__bf16)(o1[r] * inv);
    }
}

// ---------- residual + LayerNorm over AD=512 ----------
__global__ __launch_bounds__(128) void add_ln_kernel(
    const float* __restrict__ X, const float* __restrict__ T,
    const float* __restrict__ g, const float* __restrict__ b,
    float* __restrict__ Xo, __bf16* __restrict__ Xob) {
    const int row = blockIdx.x;
    const int tid = threadIdx.x;
    __shared__ float buf[AD];
    __shared__ float red[128];

    float s = 0.0f;
#pragma unroll
    for (int i = tid; i < AD; i += 128) {
        float v = X[(size_t)row * AD + i] + T[(size_t)row * AD + i];
        buf[i] = v;
        s += v;
    }
    red[tid] = s;
    __syncthreads();
    for (int off = 64; off > 0; off >>= 1) {
        if (tid < off) red[tid] += red[tid + off];
        __syncthreads();
    }
    float mean = red[0] * (1.0f / AD);
    __syncthreads();

    float s2 = 0.0f;
#pragma unroll
    for (int i = tid; i < AD; i += 128) {
        float d = buf[i] - mean;
        s2 += d * d;
    }
    red[tid] = s2;
    __syncthreads();
    for (int off = 64; off > 0; off >>= 1) {
        if (tid < off) red[tid] += red[tid + off];
        __syncthreads();
    }
    float inv = rsqrtf(red[0] * (1.0f / AD) + 1e-5f);

#pragma unroll
    for (int i = tid; i < AD; i += 128) {
        float v = (buf[i] - mean) * inv * g[i] + b[i];
        Xo[(size_t)row * AD + i] = v;
        Xob[(size_t)row * AD + i] = (__bf16)v;
    }
}

// ---------- host orchestration ----------
extern "C" void kernel_launch(void* const* d_in, const int* in_sizes, int n_in,
                              void* d_out, int out_size, void* d_ws, size_t ws_size,
                              hipStream_t stream) {
    const float* tfe    = (const float*)d_in[0];
    const int*   tids   = (const int*)d_in[2];
    const int*   ttimes = (const int*)d_in[3];
    const float* uf     = (const float*)d_in[4];
    const int*   utimes = (const int*)d_in[5];
    const int*   uids   = (const int*)d_in[6];
    const float* w2e    = (const float*)d_in[8];
    const float* e2w    = (const float*)d_in[9];
    const float* rel    = (const float*)d_in[10];
    const float* cin_w  = (const float*)d_in[11];
    const float* cin_b  = (const float*)d_in[12];
    const float* cout_w = (const float*)d_in[13];
    const float* cout_b = (const float*)d_in[14];
    const float* sin_w  = (const float*)d_in[15];
    const float* sin_b  = (const float*)d_in[16];
    const float* sout_w = (const float*)d_in[17];
    const float* sout_b = (const float*)d_in[18];
    const float* ng     = (const float*)d_in[19];
    const float* nb     = (const float*)d_in[20];
    const float* sng    = (const float*)d_in[21];
    const float* snb    = (const float*)d_in[22];
    const float* f1w    = (const float*)d_in[23];
    const float* f1b    = (const float*)d_in[24];
    const float* f2w    = (const float*)d_in[25];
    const float* f2b    = (const float*)d_in[26];
    const float* fng    = (const float*)d_in[27];
    const float* fnb    = (const float*)d_in[28];

    char* p = (char*)d_ws;
    auto carve = [&](size_t bytes) -> void* {
        void* r = (void*)p;
        p += (bytes + 255) & ~(size_t)255;
        return r;
    };
    // activations
    __bf16* trajb = (__bf16*)carve((size_t)NTOK * AD * 2);
    float*  xf    = (float*)carve((size_t)NTOK * AD * 4);
    __bf16* xb    = (__bf16*)carve((size_t)NTOK * AD * 2);
    __bf16* qb    = (__bf16*)carve((size_t)NTOK * AD * 2);
    __bf16* kb    = (__bf16*)carve((size_t)NTOK * AD * 2);
    __bf16* vb    = (__bf16*)carve((size_t)NTOK * AD * 2);
    __bf16* attnb = (__bf16*)carve((size_t)NTOK * AD * 2);
    float*  tf    = (float*)carve((size_t)NTOK * AD * 4);
    __bf16* hb    = (__bf16*)carve((size_t)NTOK * FFN * 2);
    // bf16 weights
    __bf16* wb_cin  = (__bf16*)carve((size_t)4 * 3 * AD * AD * 2);
    __bf16* wb_cout = (__bf16*)carve((size_t)4 * AD * AD * 2);
    __bf16* wb_sin  = (__bf16*)carve((size_t)3 * 3 * AD * AD * 2);
    __bf16* wb_sout = (__bf16*)carve((size_t)3 * AD * AD * 2);
    __bf16* wb_f1   = (__bf16*)carve((size_t)4 * FFN * AD * 2);
    __bf16* wb_f2   = (__bf16*)carve((size_t)4 * AD * FFN * 2);

    auto cvt = [&](const float* src, __bf16* dst, size_t n) {
        f32_to_bf16_kernel<<<(int)((n + 255) / 256), 256, 0, stream>>>(src, dst, (int)n);
    };
    cvt(cin_w,  wb_cin,  (size_t)4 * 3 * AD * AD);
    cvt(cout_w, wb_cout, (size_t)4 * AD * AD);
    cvt(sin_w,  wb_sin,  (size_t)3 * 3 * AD * AD);
    cvt(sout_w, wb_sout, (size_t)3 * AD * AD);
    cvt(f1w,    wb_f1,   (size_t)4 * FFN * AD);
    cvt(f2w,    wb_f2,   (size_t)4 * AD * FFN);

    build_inputs_kernel<<<(NTOK * AD + 255) / 256, 256, 0, stream>>>(
        tfe, tids, uf, uids, w2e, trajb, xf, xb);

    // big GEMM: M,N multiples of 64
    auto gemm = [&](const __bf16* A, int lda, const __bf16* W, int ldw, const float* bias,
                    float* Cf, __bf16* Cb, int ldc, int M, int N, int K, int act) {
        dim3 grid(M / 64, N / 64);
        gemm_wmma64_kernel<<<grid, 128, 0, stream>>>(A, lda, W, ldw, bias, Cf, Cb, ldc, K, act);
    };

    for (int l = 0; l < 4; l++) {
        if (l > 0) {
            int sl = l - 1;
            const __bf16* iw = wb_sin + (size_t)sl * 3 * AD * AD;
            const float*  ib = sin_b + (size_t)sl * 3 * AD;
            gemm(xb, AD, iw,               AD, ib,          nullptr, qb, AD, NTOK, AD, AD, 0);
            gemm(xb, AD, iw + AD * AD,     AD, ib + AD,     nullptr, kb, AD, NTOK, AD, AD, 0);
            gemm(xb, AD, iw + 2 * AD * AD, AD, ib + 2 * AD, nullptr, vb, AD, NTOK, AD, AD, 0);
            attn_fused_kernel<<<dim3(80, NH), 32, 0, stream>>>(
                qb, kb, vb, attnb, nullptr, nullptr, nullptr, 1, 2);
            gemm(attnb, AD, wb_sout + (size_t)sl * AD * AD, AD, sout_b + (size_t)sl * AD,
                 tf, nullptr, AD, NTOK, AD, AD, 0);
            add_ln_kernel<<<NTOK, 128, 0, stream>>>(
                xf, tf, sng + (size_t)sl * AD, snb + (size_t)sl * AD, xf, xb);
        }
        const __bf16* iw = wb_cin + (size_t)l * 3 * AD * AD;
        const float*  ib = cin_b + (size_t)l * 3 * AD;
        gemm(xb,    AD, iw,               AD, ib,          nullptr, qb, AD, NTOK, AD, AD, 0);
        gemm(trajb, AD, iw + AD * AD,     AD, ib + AD,     nullptr, kb, AD, NTOK, AD, AD, 0);
        gemm(trajb, AD, iw + 2 * AD * AD, AD, ib + 2 * AD, nullptr, vb, AD, NTOK, AD, AD, 0);
        attn_fused_kernel<<<dim3(80, NH), 32, 0, stream>>>(
            qb, kb, vb, attnb, utimes, ttimes, rel + (size_t)l * MAXT * NH, 0, 40);
        gemm(attnb, AD, wb_cout + (size_t)l * AD * AD, AD, cout_b + (size_t)l * AD,
             tf, nullptr, AD, NTOK, AD, AD, 0);
        add_ln_kernel<<<NTOK, 128, 0, stream>>>(
            xf, tf, ng + (size_t)l * AD, nb + (size_t)l * AD, xf, xb);
        gemm(xb, AD, wb_f1 + (size_t)l * FFN * AD, AD, f1b + (size_t)l * FFN,
             nullptr, hb, FFN, NTOK, FFN, AD, 1);
        gemm(hb, FFN, wb_f2 + (size_t)l * AD * FFN, FFN, f2b + (size_t)l * AD,
             tf, nullptr, AD, NTOK, AD, FFN, 0);
        add_ln_kernel<<<NTOK, 128, 0, stream>>>(
            xf, tf, fng + (size_t)l * AD, fnb + (size_t)l * AD, xf, xb);
    }

    // final projection: id_half (cols 256..511) @ e2w.T -> (1280 x 101) f32
    {
        dim3 grid(NTOK / 16, (VOC + 15) / 16);
        gemm_wmma_small_kernel<<<grid, 32, 0, stream>>>(
            xb + HID, AD, e2w, HID, nullptr, (float*)d_out, nullptr, VOC, VOC, HID, 0);
    }
}
